// Encoder_bound_1623497638557
// MI455X (gfx1250) — matrix operand order repaired and checked by reading, not compiled, not run on hardware
//
#include <hip/hip_runtime.h>
#include <cmath>

// Problem constants (from reference): B=4, C=512, S=8192, E=256, D=256, nb=32, W=2D=512
#define BB 4
#define CCH 512
#define SL 8192
#define EM 256
#define DW 256
#define NBW 32

typedef __attribute__((ext_vector_type(8)))  __bf16 v8bf;
typedef __attribute__((ext_vector_type(16))) __bf16 v16bf;
typedef __attribute__((ext_vector_type(8)))  float  v8f;

__device__ __forceinline__ v16bf cat8(v8bf lo, v8bf hi) {
  return __builtin_shufflevector(lo, hi, 0,1,2,3,4,5,6,7,8,9,10,11,12,13,14,15);
}
// B fragment (32x16 bf16): lane n=l%16 holds column n, K-chunk (l/16)*16..+15 contiguous.
__device__ __forceinline__ v16bf ldBfrag(const __bf16* p) {
  return cat8(*(const v8bf*)p, *(const v8bf*)(p + 8));
}
// A fragment (16x32 bf16): lane holds row l%16; K = {2*k1..2*k1+15} contiguous, k1=(l/16)*8.
__device__ __forceinline__ v16bf ldAfrag(const __bf16* row, int k1) {
  return cat8(*(const v8bf*)(row + 2 * k1), *(const v8bf*)(row + 2 * k1 + 8));
}
__device__ __forceinline__ v8f wmma_bf16(v16bf a, v16bf b, v8f c) {
  return __builtin_amdgcn_wmma_f32_16x16x32_bf16(false, a, false, b, (short)0, c, false, false);
}
// Async global->LDS 16B copy (ASYNCcnt path) + wait.  LDS byte address is the
// low 32 bits of the generic shared pointer.
__device__ __forceinline__ void async_copy_b128(void* lds, const void* g) {
  unsigned ldsAddr = (unsigned)(size_t)lds;
  asm volatile("global_load_async_to_lds_b128 %0, %1, off"
               :: "v"(ldsAddr), "v"(g) : "memory");
}
__device__ __forceinline__ void wait_async0() {
  asm volatile("s_wait_asynccnt 0x0" ::: "memory");
}

// ---------------------------------------------------------------------------
// Kernel 0: one-time weight conversion fp32 -> bf16 (enables async staging).
//   Wqkv [768][512]: rows 0..255 = Wq, 256..511 = Wk, 512..767 = Wv.
//   Wob  [512][256] = Wo.
// ---------------------------------------------------------------------------
__global__ __launch_bounds__(256) void convw_kernel(
    const float* __restrict__ Wq, const float* __restrict__ Wk,
    const float* __restrict__ Wv, const float* __restrict__ Wo,
    __bf16* __restrict__ Wqkv, __bf16* __restrict__ Wob)
{
  int i = blockIdx.x * 256 + threadIdx.x;
  if (i < 768 * 512) {
    int m = i >> 9, c = i & 511;
    const float* W = (m < 256) ? Wq : ((m < 512) ? Wk : Wv);
    Wqkv[i] = (__bf16)W[(size_t)(m & 255) * CCH + c];
  }
  if (i < 512 * 256) Wob[i] = (__bf16)Wo[i];
}

// ---------------------------------------------------------------------------
// Kernel 1: fused QKV projection.
//   m in [0,768): 0..255 -> q^T [b][s][e], 256..511 -> k^T [b][s][e],
//                 512..767 -> v [b][e][s].
// Block tile 128(M) x 128(N); K=512 streamed.  A tile staged via ASYNC DMA
// (bf16 weights), overlapped with VALU fp32->bf16 transpose staging of x1.
// ---------------------------------------------------------------------------
__global__ __launch_bounds__(256) void qkv_kernel(
    const float* __restrict__ x1,
    const __bf16* __restrict__ Wqkv,
    const float* __restrict__ bq, const float* __restrict__ bk,
    const float* __restrict__ bv,
    __bf16* __restrict__ qTb, __bf16* __restrict__ kTb, __bf16* __restrict__ vb)
{
  __shared__ __align__(16) __bf16 Ast[128 * 40];   // [m][k] stride 40 (80B rows)
  __shared__ __align__(16) __bf16 Bst[128 * 40];   // [s][k] (x1 chunk transposed)

  const int tid  = threadIdx.x;
  const int lane = tid & 31;
  const int w    = tid >> 5;
  const int s0   = blockIdx.x * 128;
  const int m0   = blockIdx.y * 128;
  const int b    = blockIdx.z;

  const int wsel = m0 >> 8;                     // 0:q 1:k 2:v
  const float* bm = (wsel == 0) ? bq : ((wsel == 1) ? bk : bv);
  const int mrow0 = m0 & 255;

  const int wm = w & 3;
  const int wn = w >> 2;
  const int k1 = (lane >> 4) * 8;
  const int kb = (lane >> 4) * 16;
  const int lm = lane & 15;

  v8f acc[2][4];
  #pragma unroll
  for (int mt = 0; mt < 2; ++mt)
    #pragma unroll
    for (int nt = 0; nt < 4; ++nt) acc[mt][nt] = {};

  for (int c0 = 0; c0 < CCH; c0 += 32) {
    // A tile (128 rows x 32 bf16 = 64B/row): async DMA, 2 x 16B per thread
    {
      const __bf16* gsrc = Wqkv + (size_t)m0 * CCH + c0;
      #pragma unroll
      for (int t = 0; t < 2; ++t) {
        int idx = tid + t * 256;                // 0..511
        int m = idx >> 2, ch = idx & 3;
        async_copy_b128(&Ast[m * 40 + ch * 8],
                        gsrc + (size_t)m * CCH + ch * 8);
      }
    }
    // B tile: x1 fp32 -> bf16 transposed staging (overlaps with async DMA)
    for (int idx = tid; idx < 32 * 128; idx += 256) {
      int c = idx >> 7, s = idx & 127;
      Bst[s * 40 + c] = (__bf16)x1[((size_t)b * CCH + c0 + c) * SL + s0 + s];
    }
    // prefetch next x1 chunk (global_prefetch_b8)
    if (c0 + 32 < CCH) {
      int c = tid >> 3, soff = (tid & 7) * 16;
      __builtin_prefetch(&x1[((size_t)b * CCH + c0 + 32 + c) * SL + s0 + soff], 0, 1);
    }
    wait_async0();
    __syncthreads();

    v16bf af[2], bf[4];
    #pragma unroll
    for (int mt = 0; mt < 2; ++mt)
      af[mt] = ldAfrag(&Ast[(wm * 32 + mt * 16 + lm) * 40], k1);
    #pragma unroll
    for (int nt = 0; nt < 4; ++nt)
      bf[nt] = ldBfrag(&Bst[(wn * 64 + nt * 16 + lm) * 40 + kb]);
    #pragma unroll
    for (int mt = 0; mt < 2; ++mt)
      #pragma unroll
      for (int nt = 0; nt < 4; ++nt)
        acc[mt][nt] = wmma_bf16(af[mt], bf[nt], acc[mt][nt]);
    __syncthreads();
  }

  #pragma unroll
  for (int mt = 0; mt < 2; ++mt)
    #pragma unroll
    for (int nt = 0; nt < 4; ++nt) {
      int nloc = wn * 64 + nt * 16 + lm;                       // s offset
      int gbase = mrow0 + wm * 32 + mt * 16 + 8 * (lane >> 4); // e base (mult 8)
      if (wsel < 2) {
        v8bf pk;
        #pragma unroll
        for (int r = 0; r < 8; ++r)
          pk[r] = (__bf16)(acc[mt][nt][r] + bm[gbase + r]);
        __bf16* dst = (wsel == 0) ? qTb : kTb;
        *(v8bf*)&dst[((size_t)b * SL + s0 + nloc) * EM + gbase] = pk;
      } else {
        #pragma unroll
        for (int r = 0; r < 8; ++r) {
          float v = acc[mt][nt][r] + bm[gbase + r];
          vb[((size_t)b * EM + gbase + r) * SL + s0 + nloc] = (__bf16)v;
        }
      }
    }
}

// ---------------------------------------------------------------------------
// Kernel 2: windowed attention + GELU.  Block = (64-query chunk, window, batch).
// Phase1: energy = q^T k, fragments loaded DIRECTLY from global q^T / k^T
// (no staging, no barriers).  Softmax in fp32 LDS with window mask.  Phase2:
// out = v . attn^T (A from global v, B from LDS attn).  GELU -> h^T [b][s][e].
// ---------------------------------------------------------------------------
#define ESM_BYTES (64 * 520 * 4)                 // energy fp32 [64][520]
#define ASM_BYTES (64 * 520 * 2)                 // attn bf16  [64][520]
#define ATTN_LDS_BYTES (ESM_BYTES + ASM_BYTES)   // 199680 B

__global__ __launch_bounds__(256) void attn_kernel(
    const __bf16* __restrict__ qTg,
    const __bf16* __restrict__ kTg,
    const __bf16* __restrict__ vg,
    const float*  __restrict__ mask,
    __bf16* __restrict__ hT)
{
  extern __shared__ __align__(16) char smem_raw[];
  float*  Esm = (float*)smem_raw;                      // [64][520]
  __bf16* Asm = (__bf16*)(smem_raw + ESM_BYTES);       // [64][520]

  const int tid  = threadIdx.x;
  const int lane = tid & 31;
  const int w    = tid >> 5;
  const int qc   = blockIdx.x;            // 0..3
  const int n    = blockIdx.y;            // 0..31
  const int b    = blockIdx.z;
  const int s0   = n * DW + qc * 64;      // first query position
  const int p0   = n * DW - 128;          // first key position (halo)
  const int lm   = lane & 15;
  const int k1   = (lane >> 4) * 8;
  const int kb   = (lane >> 4) * 16;

  const __bf16* qT = qTg + (size_t)b * SL * EM;
  const __bf16* kT = kTg + (size_t)b * SL * EM;
  const __bf16* vB = vg  + (size_t)b * EM * SL;

  // ---- Phase 1: energy[i][j], wave covers 16 i x 256 j, direct global frags
  const int iT = (w & 3) * 16;
  const int jB = (w >> 2) * 256;
  v8f acc[16];
  #pragma unroll
  for (int jt = 0; jt < 16; ++jt) acc[jt] = {};

  for (int e0 = 0; e0 < EM; e0 += 32) {
    if (e0 + 32 < EM)
      __builtin_prefetch(qT + (size_t)(s0 + (tid & 63)) * EM + e0 + 32, 0, 1);
    v16bf a = ldAfrag(qT + (size_t)(s0 + iT + lm) * EM + e0, k1);
    #pragma unroll
    for (int jt = 0; jt < 16; ++jt) {
      int j = jB + jt * 16 + lm;
      int p = p0 + j;
      p = p < 0 ? 0 : (p > SL - 1 ? SL - 1 : p);   // halo clamp (fmask zeroes it)
      v16bf bf = ldBfrag(kT + (size_t)p * EM + e0 + kb);
      acc[jt] = wmma_bf16(a, bf, acc[jt]);
    }
  }
  #pragma unroll
  for (int jt = 0; jt < 16; ++jt)
    #pragma unroll
    for (int r = 0; r < 8; ++r)
      Esm[(iT + r + 8 * (lane >> 4)) * 520 + jB + jt * 16 + lm] = acc[jt][r];
  __syncthreads();

  // ---- Softmax: 4-lane team per row, window mask -------------------------
  {
    const int row = tid >> 2;
    const int sub = tid & 3;
    float* erow = Esm + row * 520;
    const float scale = 0.0625f;          // 1/sqrt(E)
    float mx = -1e30f;
    for (int j = sub; j < 512; j += 4) {
      int p = p0 + j;
      float pm = (p >= 0 && p < SL) ? mask[(size_t)b * SL + p] : 0.0f;
      float fm = (j == 511) ? 0.0f : pm;
      float v = erow[j] * scale + __logf(fm + 1e-6f);
      erow[j] = v;
      mx = fmaxf(mx, v);
    }
    mx = fmaxf(mx, __shfl_xor(mx, 1));
    mx = fmaxf(mx, __shfl_xor(mx, 2));
    float sum = 0.0f;
    for (int j = sub; j < 512; j += 4) {
      float ex = __expf(erow[j] - mx);
      erow[j] = ex;
      sum += ex;
    }
    sum += __shfl_xor(sum, 1);
    sum += __shfl_xor(sum, 2);
    float inv = 1.0f / sum;
    for (int j = sub; j < 512; j += 4) {
      int p = p0 + j;
      float pm = (p >= 0 && p < SL) ? mask[(size_t)b * SL + p] : 0.0f;
      float fm = (j == 511) ? 0.0f : pm;
      Asm[row * 520 + j] = (__bf16)(erow[j] * inv * fm);
    }
  }
  __syncthreads();

  // ---- Phase 2: out[e][i] = sum_j v[e,p(j)] * attn[i,j], wave = 32 e x 64 i
  const int e0w = w * 32;
  v8f acc2[2][4];
  #pragma unroll
  for (int et = 0; et < 2; ++et)
    #pragma unroll
    for (int it = 0; it < 4; ++it) acc2[et][it] = {};

  for (int j0 = 0; j0 < 512; j0 += 32) {
    v16bf av[2];
    #pragma unroll
    for (int et = 0; et < 2; ++et) {
      int e = e0w + et * 16 + lm;
      const __bf16* vrow = vB + (size_t)e * SL;
      int pa = p0 + j0 + 2 * k1;          // halo: clamp; attn==0 there anyway
      int pb = pa + 8;
      pa = pa < 0 ? 0 : (pa > SL - 8 ? SL - 8 : pa);
      pb = pb < 0 ? 0 : (pb > SL - 8 ? SL - 8 : pb);
      av[et] = cat8(*(const v8bf*)(vrow + pa), *(const v8bf*)(vrow + pb));
    }
    #pragma unroll
    for (int it = 0; it < 4; ++it) {
      v16bf bf = ldBfrag(&Asm[(it * 16 + lm) * 520 + j0 + kb]);
      #pragma unroll
      for (int et = 0; et < 2; ++et)
        acc2[et][it] = wmma_bf16(av[et], bf, acc2[et][it]);
    }
  }

  // GELU (exact erf) -> h^T [b][s][e], packed 16B stores
  #pragma unroll
  for (int et = 0; et < 2; ++et)
    #pragma unroll
    for (int it = 0; it < 4; ++it) {
      int s = s0 + it * 16 + lm;
      int ebase = e0w + et * 16 + 8 * (lane >> 4);
      v8bf pk;
      #pragma unroll
      for (int r = 0; r < 8; ++r) {
        float x = acc2[et][it][r];
        pk[r] = (__bf16)(0.5f * x * (1.0f + erff(x * 0.70710678118654752f)));
      }
      *(v8bf*)&hT[((size_t)b * SL + s) * EM + ebase] = pk;
    }
}

// ---------------------------------------------------------------------------
// Kernel 3: output projection. y[b][o][s] = (Wo[o,:] . h[b,:,s] + bo[o]) * mask
// A (Wo bf16) staged once via ASYNC DMA (full K); B direct from global h^T.
// ---------------------------------------------------------------------------
#define PROJ_LDS_BYTES (128 * 264 * 2)   // Wo tile bf16 [128 m][264 e]

__global__ __launch_bounds__(256) void proj_kernel(
    const __bf16* __restrict__ hT,
    const __bf16* __restrict__ Wob,
    const float*  __restrict__ bo,
    const float*  __restrict__ mask,
    float* __restrict__ y)
{
  extern __shared__ __align__(16) char psmem[];
  __bf16* Ast = (__bf16*)psmem;          // [128][264]

  const int tid  = threadIdx.x;
  const int lane = tid & 31;
  const int w    = tid >> 5;
  const int s0   = blockIdx.x * 128;
  const int m0   = blockIdx.y * 128;
  const int b    = blockIdx.z;
  const int wm = w & 3, wn = w >> 2;
  const int k1 = (lane >> 4) * 8, kb = (lane >> 4) * 16;
  const int lm = lane & 15;

  // stage full A tile once via async DMA: 128 rows x 512B = 64KB, 16 x 16B per thread
  {
    const __bf16* gsrc = Wob + (size_t)m0 * EM;
    #pragma unroll
    for (int t = 0; t < 16; ++t) {
      int idx = tid + t * 256;            // 0..4095
      int m = idx >> 5, ch = idx & 31;
      async_copy_b128(&Ast[m * 264 + ch * 8],
                      gsrc + (size_t)m * EM + ch * 8);
    }
  }
  wait_async0();
  __syncthreads();

  v8f acc[2][4];
  #pragma unroll
  for (int mt = 0; mt < 2; ++mt)
    #pragma unroll
    for (int nt = 0; nt < 4; ++nt) acc[mt][nt] = {};

  const __bf16* hB = hT + (size_t)b * SL * EM;
  for (int c0 = 0; c0 < EM; c0 += 32) {
    if (c0 + 32 < EM)
      __builtin_prefetch(hB + (size_t)(s0 + (tid & 127)) * EM + c0 + 32, 0, 1);
    v16bf af[2], bf[4];
    #pragma unroll
    for (int mt = 0; mt < 2; ++mt)
      af[mt] = ldAfrag(&Ast[(wm * 32 + mt * 16 + lm) * 264 + c0], k1);
    #pragma unroll
    for (int nt = 0; nt < 4; ++nt)
      bf[nt] = ldBfrag(hB + (size_t)(s0 + wn * 64 + nt * 16 + lm) * EM + c0 + kb);
    #pragma unroll
    for (int mt = 0; mt < 2; ++mt)
      #pragma unroll
      for (int nt = 0; nt < 4; ++nt)
        acc[mt][nt] = wmma_bf16(af[mt], bf[nt], acc[mt][nt]);
  }

  #pragma unroll
  for (int mt = 0; mt < 2; ++mt)
    #pragma unroll
    for (int nt = 0; nt < 4; ++nt) {
      int nloc = wn * 64 + nt * 16 + lm;
      int s = s0 + nloc;
      float mv = mask[(size_t)b * SL + s];
      #pragma unroll
      for (int r = 0; r < 8; ++r) {
        int mloc = wm * 32 + mt * 16 + r + 8 * (lane >> 4);
        int o = m0 + mloc;
        float val = (acc[mt][nt][r] + bo[o]) * mv;
        y[((size_t)b * CCH + o) * SL + s] = val;
      }
    }
}

// ---------------------------------------------------------------------------
extern "C" void kernel_launch(void* const* d_in, const int* in_sizes, int n_in,
                              void* d_out, int out_size, void* d_ws, size_t ws_size,
                              hipStream_t stream) {
  const float* x1   = (const float*)d_in[0];
  const float* mask = (const float*)d_in[1];
  const float* Wq   = (const float*)d_in[2];
  const float* bq   = (const float*)d_in[3];
  const float* Wk   = (const float*)d_in[4];
  const float* bk   = (const float*)d_in[5];
  const float* Wv   = (const float*)d_in[6];
  const float* bv   = (const float*)d_in[7];
  const float* Wo   = (const float*)d_in[8];
  const float* bo   = (const float*)d_in[9];
  float* y = (float*)d_out;

  // Workspace (bf16): qT | kT | v | hT (each [B][S][E]-sized, 16 MiB) then
  // Wqkv [768][512] + Wob [512][256] (1 MiB).  Total ~65 MiB.
  const size_t plane = (size_t)BB * SL * EM;
  __bf16* qT   = (__bf16*)d_ws;
  __bf16* kT   = qT + plane;
  __bf16* vb   = kT + plane;
  __bf16* hT   = vb + plane;
  __bf16* Wqkv = hT + plane;
  __bf16* Wob  = Wqkv + (size_t)768 * CCH;

  hipFuncSetAttribute(reinterpret_cast<const void*>(attn_kernel),
                      hipFuncAttributeMaxDynamicSharedMemorySize,
                      (int)ATTN_LDS_BYTES);
  hipFuncSetAttribute(reinterpret_cast<const void*>(proj_kernel),
                      hipFuncAttributeMaxDynamicSharedMemorySize,
                      (int)PROJ_LDS_BYTES);

  convw_kernel<<<dim3((768 * 512 + 255) / 256), 256, 0, stream>>>(
      Wq, Wk, Wv, Wo, Wqkv, Wob);
  qkv_kernel<<<dim3(SL / 128, 768 / 128, BB), 256, 0, stream>>>(
      x1, Wqkv, bq, bk, bv, qT, kT, vb);
  attn_kernel<<<dim3(4, NBW, BB), 256, ATTN_LDS_BYTES, stream>>>(
      qT, kT, vb, mask, hT);
  proj_kernel<<<dim3(SL / 128, CCH / 128, BB), 256, PROJ_LDS_BYTES, stream>>>(
      hT, Wob, bo, mask, y);
}
